// MSGMVC_17746804867485
// MI455X (gfx1250) — matrix operand order repaired and checked by reading, not compiled, not run on hardware
//
#include <hip/hip_runtime.h>
#include <hip/hip_bf16.h>

// ---------------------------------------------------------------------------
// Problem constants (match reference)
// ---------------------------------------------------------------------------
#define BATCH   8192
#define D1      2000
#define D2      1500
#define TRUNKD  1024
#define HEADD   512
#define ZCD     256
#define ZSD     256
#define NCENT   10
#define TOPK    10
#define EPSV    1e-8f

// GEMM tiling: 128x64 tile, K-step 32, 8 waves (wave32), double-buffered LDS
#define BM 128
#define BN 64
#define BK 32
#define LDSB 40   // halves per LDS row: 80B stride -> 16B aligned, conflict-free frags

typedef __attribute__((ext_vector_type(16))) __bf16 v16bf;
typedef __attribute__((ext_vector_type(8)))  float  v8f;

union FragBF {
    v16bf v;
    unsigned int u[8];
};
union StA { uint2 q[4]; unsigned short h[16]; };   // 16 bf16 per thread (A tail)
union StB { uint2 q[2]; unsigned short h[8];  };   // 8  bf16 per thread (B tile)

__device__ __forceinline__ unsigned short f2bf(float f) {
    unsigned int u = __builtin_bit_cast(unsigned int, f);
    unsigned int r = u + 0x7FFFu + ((u >> 16) & 1u);   // round-to-nearest-even
    return (unsigned short)(r >> 16);
}

// Async copy 8 bytes global -> LDS (CDNA5 GLOBAL_LOAD_ASYNC_TO_LDS_B64, ASYNCcnt).
// lds_off: LDS byte address (low 32 bits of the flat shared pointer per ISA 10.2).
__device__ __forceinline__ void async_lds_b64(unsigned lds_off, const void* gptr) {
    asm volatile("global_load_async_to_lds_b64 %0, %1, off"
                 :: "v"(lds_off), "v"((unsigned long long)(size_t)gptr)
                 : "memory");
}
__device__ __forceinline__ void wait_async0() {
    asm volatile("s_wait_asynccnt 0" ::: "memory");
}

// ---------------------------------------------------------------------------
// BF16-in / f32-accumulate WMMA GEMM, specialized on TRANSB.
//   A  : bf16 [M,K] row-major          (M must be a multiple of 128 -- true here)
//   W  : bf16 [K,N] (TRANSB=0) or [N,K] (TRANSB=1) row-major
//   C  : optional f32 out (ld = ldc), bias+relu applied
//   Cb : optional bf16 out (ld = ldcb), same value, for chaining into next GEMM
// Pipeline: prologue stage -> branch-free steady loop over full K tiles ->
//           peeled last-full-tile + optional guarded K-tail.
// ---------------------------------------------------------------------------
template <int TRANSB>
__global__ __launch_bounds__(256)
void msgmvc_gemm_bf16(const unsigned short* __restrict__ A,
                      const unsigned short* __restrict__ W,
                      const float* __restrict__ bias,
                      float* __restrict__ C, unsigned short* __restrict__ Cb,
                      int M, int N, int K, int relu, int ldc, int ldcb)
{
    __shared__ __align__(16) unsigned short As[2][BM][LDSB];
    __shared__ __align__(16) unsigned short Bs[2][BN][LDSB];

    const int tid  = threadIdx.x;
    const int lane = tid & 31;
    const int wid  = tid >> 5;                 // 0..7
    const int m0   = blockIdx.y * BM;
    const int n0   = blockIdx.x * BN;
    const int wm   = (wid & 3) * 32;           // wave covers rows wm..wm+31
    const int wn   = (wid >> 2) * 32;          // wave covers cols wn..wn+31

    // staging coordinates (uniform per thread across the whole K loop)
    const int ar = tid >> 1;                   // A: row 0..127
    const int ac = (tid & 1) * 16;             // A: col 0 or 16
    const int bn_t = TRANSB ? (tid >> 2) : (tid & 63);      // B: n row in LDS
    const int bk_t = TRANSB ? ((tid & 3) * 8) : ((tid >> 6) * 8);
    const bool bok = (n0 + bn_t) < N;

    v8f acc00 = {}, acc01 = {}, acc10 = {}, acc11 = {};

    StA ra; StB rb;
    const int  nf   = K / BK;                  // full tiles (K >= 256 always)
    const bool tail = (K % BK) != 0;

    // ---- full-tile staging (async for A, and for B when TRANSB) ----
    auto stageFullA = [&](int buf, int k0) {
        const unsigned short* asrc = A + (size_t)(m0 + ar) * K + k0 + ac;
        #pragma unroll
        for (int j = 0; j < 4; ++j)
            async_lds_b64((unsigned)(size_t)&As[buf][ar][ac + 4 * j], asrc + 4 * j);
    };
    auto stageFullB = [&](int buf, int k0) {
        if (TRANSB) {
            // OOB-n columns never stored in epilogue -> stale LDS harmless
            if (bok) {
                const unsigned short* bsrc = W + (size_t)(n0 + bn_t) * K + k0 + bk_t;
                async_lds_b64((unsigned)(size_t)&Bs[buf][bn_t][bk_t], bsrc);
                async_lds_b64((unsigned)(size_t)&Bs[buf][bn_t][bk_t + 4], bsrc + 4);
            }
        } else {
            // k-strided u16 gathers: register path (no 16-bit async op)
            const int gn = n0 + bn_t;
            #pragma unroll
            for (int i = 0; i < 8; ++i)
                rb.h[i] = bok ? W[(size_t)(k0 + bk_t + i) * N + gn] : (unsigned short)0;
        }
    };
    // ---- guarded K-tail staging into registers ----
    auto stageTail = [&](int k0) {
        const unsigned short* asrc = A + (size_t)(m0 + ar) * K + k0 + ac;
        #pragma unroll
        for (int i = 0; i < 16; ++i) {
            int gk = k0 + ac + i;
            ra.h[i] = (gk < K) ? asrc[i] : (unsigned short)0;
        }
        if (TRANSB) {
            const unsigned short* bsrc = W + (size_t)(n0 + bn_t) * K + k0 + bk_t;
            #pragma unroll
            for (int i = 0; i < 8; ++i) {
                int gk = k0 + bk_t + i;
                rb.h[i] = (bok && gk < K) ? bsrc[i] : (unsigned short)0;
            }
        } else {
            const int gn = n0 + bn_t;
            #pragma unroll
            for (int i = 0; i < 8; ++i) {
                int gk = k0 + bk_t + i;
                rb.h[i] = (bok && gk < K) ? W[(size_t)gk * N + gn] : (unsigned short)0;
            }
        }
    };
    auto flushA = [&](int buf) {
        #pragma unroll
        for (int j = 0; j < 4; ++j) *(uint2*)&As[buf][ar][ac + 4 * j] = ra.q[j];
    };
    auto flushB = [&](int buf) {
        *(uint2*)&Bs[buf][bn_t][bk_t]     = rb.q[0];
        *(uint2*)&Bs[buf][bn_t][bk_t + 4] = rb.q[1];
    };

    // ---- fragment loads + 4 WMMAs on one LDS buffer (ISA 7.12.2 layouts) ----
    auto compute = [&](int cur) {
        FragBF a0, a1, b0, b1;
        const int mr  = lane & 15;
        const int kba = (lane & 16) ? 8 : 0;
        #pragma unroll
        for (int j = 0; j < 8; ++j) {
            const int kk = (j < 4) ? (kba + 2 * j) : (16 + kba + 2 * (j - 4));
            a0.u[j] = *(const unsigned int*)&As[cur][wm + mr][kk];
            a1.u[j] = *(const unsigned int*)&As[cur][wm + 16 + mr][kk];
        }
        const int kbb = (lane & 16) ? 8 : 0;
        #pragma unroll
        for (int j = 0; j < 8; ++j) {
            const int kk = (j < 4) ? (kbb + 2 * j) : (16 + kbb + 2 * (j - 4));
            b0.u[j] = *(const unsigned int*)&Bs[cur][wn + mr][kk];
            b1.u[j] = *(const unsigned int*)&Bs[cur][wn + 16 + mr][kk];
        }
        acc00 = __builtin_amdgcn_wmma_f32_16x16x32_bf16(false, a0.v, false, b0.v, (short)0, acc00, false, false);
        acc01 = __builtin_amdgcn_wmma_f32_16x16x32_bf16(false, a0.v, false, b1.v, (short)0, acc01, false, false);
        acc10 = __builtin_amdgcn_wmma_f32_16x16x32_bf16(false, a1.v, false, b0.v, (short)0, acc10, false, false);
        acc11 = __builtin_amdgcn_wmma_f32_16x16x32_bf16(false, a1.v, false, b1.v, (short)0, acc11, false, false);
    };

    // ---- prologue: stage tile 0 (always a full tile) ----
    stageFullA(0, 0);
    stageFullB(0, 0);
    if (!TRANSB) flushB(0);
    wait_async0();
    __syncthreads();

    // ---- steady state: branch-free loop over full tiles 0..nf-2 ----
    for (int kt = 0; kt + 1 < nf; ++kt) {
        const int cur = kt & 1;
        stageFullA(cur ^ 1, (kt + 1) * BK);
        stageFullB(cur ^ 1, (kt + 1) * BK);
        compute(cur);
        if (!TRANSB) flushB(cur ^ 1);          // deferred weight-tile store
        wait_async0();
        __syncthreads();
    }

    // ---- last full tile + optional K-tail ----
    {
        const int cur = (nf - 1) & 1;
        if (tail) stageTail(nf * BK);
        compute(cur);
        if (tail) {
            flushA(cur ^ 1);
            flushB(cur ^ 1);
            __syncthreads();
            compute(cur ^ 1);
        }
    }

    // ---- epilogue: C layout (VGPR r -> M=r | r+8 by lane half, N = lane&15) ----
    const int mbase = m0 + wm + ((lane & 16) ? 8 : 0);
    const int nc0   = n0 + wn + (lane & 15);
    const int nc1   = nc0 + 16;
    auto emit = [&](const v8f& acc, int gmb, int nc) {
        if (nc >= N) return;
        const float bv = bias ? bias[nc] : 0.0f;
        #pragma unroll
        for (int r = 0; r < 8; ++r) {
            float v = acc[r] + bv;
            if (relu) v = fmaxf(v, 0.0f);
            const int gm = gmb + r;            // M multiple of 128 -> always in range
            if (C)  C[(size_t)gm * ldc + nc] = v;
            if (Cb) Cb[(size_t)gm * ldcb + nc] = f2bf(v);
        }
    };
    emit(acc00, mbase,      nc0);
    emit(acc01, mbase,      nc1);
    emit(acc10, mbase + 16, nc0);
    emit(acc11, mbase + 16, nc1);
}

// ---------------------------------------------------------------------------
// f32 -> bf16 bulk conversion (n multiple of 4)
// ---------------------------------------------------------------------------
__global__ void msgmvc_cvt_bf16(const float* __restrict__ src,
                                unsigned short* __restrict__ dst, size_t n)
{
    size_t stride = (size_t)gridDim.x * blockDim.x * 4;
    for (size_t i = ((size_t)blockIdx.x * blockDim.x + threadIdx.x) * 4; i < n; i += stride) {
        float4 f = *(const float4*)(src + i);
        uint2 u;
        u.x = (unsigned int)f2bf(f.x) | ((unsigned int)f2bf(f.y) << 16);
        u.y = (unsigned int)f2bf(f.z) | ((unsigned int)f2bf(f.w) << 16);
        *(uint2*)(dst + i) = u;
    }
}

// x -> bf16 with missing rows zeroed (D multiple of 4)
__global__ void msgmvc_cvt_x(const float* __restrict__ x, const int* __restrict__ pres,
                             unsigned short* __restrict__ dst, int Bn, int D)
{
    size_t total = (size_t)Bn * D;
    size_t stride = (size_t)gridDim.x * blockDim.x * 4;
    for (size_t i = ((size_t)blockIdx.x * blockDim.x + threadIdx.x) * 4; i < total; i += stride) {
        int b = (int)(i / D);
        bool ok = pres[b] != 0;
        float4 f = *(const float4*)(x + i);
        if (!ok) { f.x = f.y = f.z = f.w = 0.f; }
        uint2 u;
        u.x = (unsigned int)f2bf(f.x) | ((unsigned int)f2bf(f.y) << 16);
        u.y = (unsigned int)f2bf(f.z) | ((unsigned int)f2bf(f.w) << 16);
        *(uint2*)(dst + i) = u;
    }
}

// ---------------------------------------------------------------------------
// out[b,d] = x[b,d] * (p0[b] && p1[b])
// ---------------------------------------------------------------------------
__global__ void msgmvc_keep_mask(const float* __restrict__ x,
                                 const int* __restrict__ p0, const int* __restrict__ p1,
                                 float* __restrict__ out, int Bn, int D)
{
    size_t total = (size_t)Bn * D;
    for (size_t i = (size_t)blockIdx.x * blockDim.x + threadIdx.x; i < total;
         i += (size_t)gridDim.x * blockDim.x) {
        int b = (int)(i / D);
        float keep = (p0[b] != 0 && p1[b] != 0) ? 1.0f : 0.0f;
        out[i] = x[i] * keep;
    }
}

// ---------------------------------------------------------------------------
// Masked per-column mean / std over the batch (one block per column)
// ---------------------------------------------------------------------------
__global__ __launch_bounds__(256)
void msgmvc_colstats(const float* __restrict__ X, const int* __restrict__ pres,
                     float* __restrict__ mu, float* __restrict__ sd, int Bn, int D)
{
    __shared__ float s1[256], s2[256], s3[256];
    const int d = blockIdx.x;
    const int t = threadIdx.x;
    float s = 0.f, sq = 0.f, c = 0.f;
    for (int b = t; b < Bn; b += 256) {
        if (pres[b]) {
            float v = X[(size_t)b * D + d];
            s += v; sq += v * v; c += 1.f;
        }
    }
    s1[t] = s; s2[t] = sq; s3[t] = c;
    __syncthreads();
    for (int o = 128; o > 0; o >>= 1) {
        if (t < o) { s1[t] += s1[t + o]; s2[t] += s2[t + o]; s3[t] += s3[t + o]; }
        __syncthreads();
    }
    if (t == 0) {
        float cnt = fmaxf(s3[0], 1.0f);
        float m   = s1[0] / cnt;
        float var = fmaxf(s2[0] / cnt - m * m, 0.0f);
        mu[d] = m;
        sd[d] = fmaxf(sqrtf(var), EPSV);
    }
}

// ---------------------------------------------------------------------------
// Cross-view content fill (V=2 closed form), emits f32 output + bf16 mirror
// ---------------------------------------------------------------------------
__global__ void msgmvc_fill_content(const float* __restrict__ z0, const float* __restrict__ z1,
                                    const int* __restrict__ p0, const int* __restrict__ p1,
                                    const float* __restrict__ mu0, const float* __restrict__ sd0,
                                    const float* __restrict__ mu1, const float* __restrict__ sd1,
                                    float* __restrict__ out0, float* __restrict__ out1,
                                    unsigned short* __restrict__ ob0, unsigned short* __restrict__ ob1,
                                    int Bn, int D)
{
    size_t total = (size_t)Bn * D;
    for (size_t i = (size_t)blockIdx.x * blockDim.x + threadIdx.x; i < total;
         i += (size_t)gridDim.x * blockDim.x) {
        int b = (int)(i / D);
        int d = (int)(i % D);
        bool q0 = p0[b] != 0, q1 = p1[b] != 0;
        float xn0 = (z0[i] - mu0[d]) / sd0[d];
        float xn1 = (z1[i] - mu1[d]) / sd1[d];
        float v0 = q0 ? z0[i] : ((q1 ? xn1 : 0.f) * sd0[d] + mu0[d]);
        float v1 = q1 ? z1[i] : ((q0 ? xn0 : 0.f) * sd1[d] + mu1[d]);
        out0[i] = v0; out1[i] = v1;
        ob0[i] = f2bf(v0); ob1[i] = f2bf(v1);
    }
}

// ---------------------------------------------------------------------------
// Row squared norms (one wave per row)
// ---------------------------------------------------------------------------
__global__ __launch_bounds__(256)
void msgmvc_sqnorm(const float* __restrict__ Z, float* __restrict__ sq, int Bn, int D)
{
    int b = blockIdx.x * 8 + (threadIdx.x >> 5);
    int lane = threadIdx.x & 31;
    if (b >= Bn) return;
    float s = 0.f;
    for (int d = lane; d < D; d += 32) {
        float v = Z[(size_t)b * D + d];
        s += v * v;
    }
    for (int o = 16; o > 0; o >>= 1) s += __shfl_xor(s, o, 32);
    if (lane == 0) sq[b] = s;
}

// ---------------------------------------------------------------------------
// Per-row top-K nearest neighbors from Gram matrix (block per row)
// ---------------------------------------------------------------------------
__global__ __launch_bounds__(256)
void msgmvc_topk(const float* __restrict__ G, const float* __restrict__ sq,
                 const int* __restrict__ pres, int* __restrict__ idx, int Bn)
{
    __shared__ float svals[256 * TOPK];
    __shared__ int   sidx[256 * TOPK];
    const int i = blockIdx.x;
    const int t = threadIdx.x;
    float bd[TOPK]; int bi[TOPK];
    #pragma unroll
    for (int j = 0; j < TOPK; ++j) { bd[j] = __builtin_inff(); bi[j] = 0; }
    const float sqi = sq[i];
    for (int j = t; j < Bn; j += 256) {
        float d2 = (pres[j] != 0 && j != i)
                       ? fmaxf(sqi + sq[j] - 2.0f * G[(size_t)i * Bn + j], 0.0f)
                       : __builtin_inff();
        if (d2 < bd[TOPK - 1]) {
            int pos = TOPK - 1;
            while (pos > 0 && bd[pos - 1] > d2) {
                bd[pos] = bd[pos - 1]; bi[pos] = bi[pos - 1]; --pos;
            }
            bd[pos] = d2; bi[pos] = j;
        }
    }
    #pragma unroll
    for (int j = 0; j < TOPK; ++j) { svals[t * TOPK + j] = bd[j]; sidx[t * TOPK + j] = bi[j]; }
    __syncthreads();
    for (int stride = 128; stride >= 1; stride >>= 1) {
        if (t < stride) {
            float mv[TOPK]; int mi[TOPK];
            int pa = 0, pb = 0;
            #pragma unroll
            for (int j = 0; j < TOPK; ++j) {
                float va = svals[t * TOPK + pa];
                float vb = svals[(t + stride) * TOPK + pb];
                if (va <= vb) { mv[j] = va; mi[j] = sidx[t * TOPK + pa]; ++pa; }
                else          { mv[j] = vb; mi[j] = sidx[(t + stride) * TOPK + pb]; ++pb; }
            }
            #pragma unroll
            for (int j = 0; j < TOPK; ++j) { svals[t * TOPK + j] = mv[j]; sidx[t * TOPK + j] = mi[j]; }
        }
        __syncthreads();
    }
    if (t < TOPK) idx[(size_t)i * TOPK + t] = sidx[t];
}

// ---------------------------------------------------------------------------
// Style fill: present rows keep zs, missing rows take mean of K neighbors
// ---------------------------------------------------------------------------
__global__ void msgmvc_fill_style(const float* __restrict__ zs, const int* __restrict__ pres,
                                  const int* __restrict__ idx, float* __restrict__ out,
                                  int Bn, int D)
{
    size_t total = (size_t)Bn * D;
    for (size_t i = (size_t)blockIdx.x * blockDim.x + threadIdx.x; i < total;
         i += (size_t)gridDim.x * blockDim.x) {
        int b = (int)(i / D);
        int d = (int)(i % D);
        if (pres[b] != 0) {
            out[i] = zs[i];
        } else {
            float s = 0.f;
            #pragma unroll
            for (int j = 0; j < TOPK; ++j)
                s += zs[(size_t)idx[b * TOPK + j] * D + d];
            out[i] = s * (1.0f / TOPK);
        }
    }
}

// ---------------------------------------------------------------------------
// DEC student-t soft assignment (one wave per row)
// ---------------------------------------------------------------------------
__global__ __launch_bounds__(256)
void msgmvc_qsoft(const float* __restrict__ zc, const float* __restrict__ centers,
                  float* __restrict__ q, int Bn, int D)
{
    int b = blockIdx.x * 8 + (threadIdx.x >> 5);
    int lane = threadIdx.x & 31;
    if (b >= Bn) return;
    float zl[8];   // D == 256 -> 8 per lane
    #pragma unroll
    for (int j = 0; j < 8; ++j) zl[j] = zc[(size_t)b * D + lane + j * 32];
    float qv[NCENT];
    float qsum = 0.f;
    #pragma unroll
    for (int c = 0; c < NCENT; ++c) {
        float s = 0.f;
        #pragma unroll
        for (int j = 0; j < 8; ++j) {
            float dlt = zl[j] - centers[(size_t)c * D + lane + j * 32];
            s += dlt * dlt;
        }
        for (int o = 16; o > 0; o >>= 1) s += __shfl_xor(s, o, 32);
        float qq = 1.0f / (1.0f + s);
        qv[c] = qq; qsum += qq;
    }
    if (lane == 0) {
        float inv = 1.0f / qsum;
        #pragma unroll
        for (int c = 0; c < NCENT; ++c) q[(size_t)b * NCENT + c] = qv[c] * inv;
    }
}

// ---------------------------------------------------------------------------
// Host-side launch orchestration
// ---------------------------------------------------------------------------
static inline void launch_gemm(const unsigned short* A, const unsigned short* W,
                               const float* bias, float* C, unsigned short* Cb,
                               int M, int N, int K, int transB, int relu,
                               int ldc, int ldcb, hipStream_t stream)
{
    dim3 grid((N + BN - 1) / BN, (M + BM - 1) / BM);
    if (transB)
        msgmvc_gemm_bf16<1><<<grid, 256, 0, stream>>>(A, W, bias, C, Cb, M, N, K, relu, ldc, ldcb);
    else
        msgmvc_gemm_bf16<0><<<grid, 256, 0, stream>>>(A, W, bias, C, Cb, M, N, K, relu, ldc, ldcb);
}

extern "C" void kernel_launch(void* const* d_in, const int* in_sizes, int n_in,
                              void* d_out, int out_size, void* d_ws, size_t ws_size,
                              hipStream_t stream)
{
    (void)in_sizes; (void)n_in; (void)out_size; (void)ws_size;

    const float* x1   = (const float*)d_in[0];
    const float* x2   = (const float*)d_in[1];
    const int*   mask = (const int*)d_in[2];
    const int*   p0   = mask;
    const int*   p1   = mask + BATCH;
    const int*   pv[2] = { p0, p1 };
    const int    Dv[2] = { D1, D2 };

    const float* fin[50];
    for (int i = 0; i < 50; ++i) fin[i] = (const float*)d_in[i];

    // output layout (floats, concatenated in return order)
    float* out = (float*)d_out;
    float* o_x1 = out;
    float* o_x2 = o_x1 + (size_t)BATCH * D1;
    float* o_zc[2]; o_zc[0] = o_x2 + (size_t)BATCH * D2;
    o_zc[1] = o_zc[0] + (size_t)BATCH * ZCD;
    float* o_zs[2]; o_zs[0] = o_zc[1] + (size_t)BATCH * ZCD;
    o_zs[1] = o_zs[0] + (size_t)BATCH * ZSD;
    float* o_rc[2]; o_rc[0] = o_zs[1] + (size_t)BATCH * ZSD;
    o_rc[1] = o_rc[0] + (size_t)BATCH * D1;
    float* o_q[2];  o_q[0]  = o_rc[1] + (size_t)BATCH * D2;
    o_q[1]  = o_q[0] + (size_t)BATCH * NCENT;

    // ---------------- workspace carve-up ----------------
    char* wsp = (char*)d_ws;
    auto walloc = [&](size_t bytes) -> void* {
        char* p = wsp;
        wsp += (bytes + 255) & ~(size_t)255;
        return (void*)p;
    };
    typedef unsigned short ush;

    // bf16 weight mirrors (converted once per launch)
    const int widx[22]  = { 4, 6, 8, 10, 12, 14, 16, 18, 20, 22, 24, 26,
                            28, 30, 32, 34, 36, 38, 40, 42, 44, 46 };
    const size_t welem[22] = {
        (size_t)D1 * TRUNKD, (size_t)TRUNKD * TRUNKD,
        (size_t)D2 * TRUNKD, (size_t)TRUNKD * TRUNKD,
        (size_t)TRUNKD * HEADD, (size_t)HEADD * ZCD,
        (size_t)TRUNKD * HEADD, (size_t)HEADD * ZCD,
        (size_t)TRUNKD * HEADD, (size_t)HEADD * ZSD,
        (size_t)TRUNKD * HEADD, (size_t)HEADD * ZSD,
        (size_t)ZCD * HEADD, (size_t)ZCD * HEADD,
        (size_t)ZSD * HEADD, (size_t)ZSD * HEADD,
        (size_t)TRUNKD * TRUNKD, (size_t)TRUNKD * TRUNKD, (size_t)TRUNKD * D1,
        (size_t)TRUNKD * TRUNKD, (size_t)TRUNKD * TRUNKD, (size_t)TRUNKD * D2 };
    ush* wb[50] = {};
    for (int i = 0; i < 22; ++i) wb[widx[i]] = (ush*)walloc(welem[i] * 2);

    ush* xb[2];  xb[0] = (ush*)walloc((size_t)BATCH * D1 * 2);
    xb[1] = (ush*)walloc((size_t)BATCH * D2 * 2);
    ush* tAb   = (ush*)walloc((size_t)BATCH * TRUNKD * 2);
    ush* tBb   = (ush*)walloc((size_t)BATCH * TRUNKD * 2);
    ush* z1b   = (ush*)walloc((size_t)BATCH * TRUNKD * 2);
    ush* h512  = (ush*)walloc((size_t)BATCH * HEADD * 2);
    ush* catb  = (ush*)walloc((size_t)BATCH * 2 * HEADD * 2);
    ush* zcb[2]; zcb[0] = (ush*)walloc((size_t)BATCH * ZCD * 2);
    zcb[1] = (ush*)walloc((size_t)BATCH * ZCD * 2);
    ush* zsb[2]; zsb[0] = (ush*)walloc((size_t)BATCH * ZSD * 2);
    zsb[1] = (ush*)walloc((size_t)BATCH * ZSD * 2);
    ush* zcfb[2]; zcfb[0] = (ush*)walloc((size_t)BATCH * ZCD * 2);
    zcfb[1] = (ush*)walloc((size_t)BATCH * ZCD * 2);
    float* zcf[2]; zcf[0] = (float*)walloc((size_t)BATCH * ZCD * 4);
    zcf[1] = (float*)walloc((size_t)BATCH * ZCD * 4);
    float* zsf[2]; zsf[0] = (float*)walloc((size_t)BATCH * ZSD * 4);
    zsf[1] = (float*)walloc((size_t)BATCH * ZSD * 4);
    float* mu[2];  mu[0] = (float*)walloc(ZCD * 4);  mu[1] = (float*)walloc(ZCD * 4);
    float* sd[2];  sd[0] = (float*)walloc(ZCD * 4);  sd[1] = (float*)walloc(ZCD * 4);
    float* sqb  = (float*)walloc((size_t)BATCH * 4);
    int*   idxb = (int*)walloc((size_t)BATCH * TOPK * 4);
    float* G    = (float*)walloc((size_t)BATCH * BATCH * 4);

    const int EW_BLOCKS = 2048;

    // ---------------- precision conversion pass (once per launch) ----------------
    for (int i = 0; i < 22; ++i)
        msgmvc_cvt_bf16<<<512, 256, 0, stream>>>(fin[widx[i]], wb[widx[i]], welem[i]);
    msgmvc_cvt_x<<<EW_BLOCKS, 256, 0, stream>>>(x1, p0, xb[0], BATCH, D1);
    msgmvc_cvt_x<<<EW_BLOCKS, 256, 0, stream>>>(x2, p1, xb[1], BATCH, D2);

    // ---------------- encoder / decoder stacks (per view) ----------------
    for (int v = 0; v < 2; ++v) {
        const int et = 4 + 4 * v, ec = 12 + 4 * v, es = 20 + 4 * v;
        const int dc = 28 + 2 * v, ds = 32 + 2 * v, dt = 36 + 6 * v;
        // enc_trunk
        launch_gemm(xb[v], wb[et], fin[et + 1], nullptr, tAb, BATCH, TRUNKD, Dv[v], 0, 1, 0, TRUNKD, stream);
        launch_gemm(tAb, wb[et + 2], fin[et + 3], nullptr, z1b, BATCH, TRUNKD, TRUNKD, 0, 0, 0, TRUNKD, stream);
        // enc_content
        launch_gemm(z1b, wb[ec], fin[ec + 1], nullptr, h512, BATCH, HEADD, TRUNKD, 0, 1, 0, HEADD, stream);
        launch_gemm(h512, wb[ec + 2], fin[ec + 3], zcf[v], zcb[v], BATCH, ZCD, HEADD, 0, 0, ZCD, ZCD, stream);
        // enc_style
        launch_gemm(z1b, wb[es], fin[es + 1], nullptr, h512, BATCH, HEADD, TRUNKD, 0, 1, 0, HEADD, stream);
        launch_gemm(h512, wb[es + 2], fin[es + 3], zsf[v], zsb[v], BATCH, ZSD, HEADD, 0, 0, ZSD, ZSD, stream);
        // dec_content / dec_style -> concat halves via ldcb
        launch_gemm(zcb[v], wb[dc], fin[dc + 1], nullptr, catb,         BATCH, HEADD, ZCD, 0, 0, 0, 2 * HEADD, stream);
        launch_gemm(zsb[v], wb[ds], fin[ds + 1], nullptr, catb + HEADD, BATCH, HEADD, ZSD, 0, 0, 0, 2 * HEADD, stream);
        // dec_trunk -> recon straight into d_out (f32)
        launch_gemm(catb, wb[dt], fin[dt + 1], nullptr, tAb, BATCH, TRUNKD, 2 * HEADD, 0, 1, 0, TRUNKD, stream);
        launch_gemm(tAb, wb[dt + 2], fin[dt + 3], nullptr, tBb, BATCH, TRUNKD, TRUNKD, 0, 1, 0, TRUNKD, stream);
        launch_gemm(tBb, wb[dt + 4], fin[dt + 5], o_rc[v], nullptr, BATCH, Dv[v], TRUNKD, 0, 0, Dv[v], 0, stream);
    }

    // ---------------- content fill ----------------
    msgmvc_colstats<<<ZCD, 256, 0, stream>>>(zcf[0], p0, mu[0], sd[0], BATCH, ZCD);
    msgmvc_colstats<<<ZCD, 256, 0, stream>>>(zcf[1], p1, mu[1], sd[1], BATCH, ZCD);
    msgmvc_fill_content<<<EW_BLOCKS, 256, 0, stream>>>(zcf[0], zcf[1], p0, p1,
                                                       mu[0], sd[0], mu[1], sd[1],
                                                       o_zc[0], o_zc[1], zcfb[0], zcfb[1],
                                                       BATCH, ZCD);

    // ---------------- style fill (KNN in content space) + q ----------------
    for (int v = 0; v < 2; ++v) {
        msgmvc_sqnorm<<<BATCH / 8, 256, 0, stream>>>(o_zc[v], sqb, BATCH, ZCD);
        launch_gemm(zcfb[v], zcfb[v], nullptr, G, nullptr, BATCH, BATCH, ZCD, 1, 0, BATCH, 0, stream);
        msgmvc_topk<<<BATCH, 256, 0, stream>>>(G, sqb, pv[v], idxb, BATCH);
        msgmvc_fill_style<<<EW_BLOCKS, 256, 0, stream>>>(zsf[v], pv[v], idxb, o_zs[v], BATCH, ZSD);
        msgmvc_qsoft<<<BATCH / 8, 256, 0, stream>>>(o_zc[v], fin[48 + v], o_q[v], BATCH, ZCD);
    }

    // ---------------- keep-masked passthrough and recon ----------------
    msgmvc_keep_mask<<<EW_BLOCKS, 256, 0, stream>>>(x1, p0, p1, o_x1, BATCH, D1);
    msgmvc_keep_mask<<<EW_BLOCKS, 256, 0, stream>>>(x2, p0, p1, o_x2, BATCH, D2);
    msgmvc_keep_mask<<<EW_BLOCKS, 256, 0, stream>>>(o_rc[0], p0, p1, o_rc[0], BATCH, D1);
    msgmvc_keep_mask<<<EW_BLOCKS, 256, 0, stream>>>(o_rc[1], p0, p1, o_rc[1], BATCH, D2);
}
